// MCB_42949672960828
// MI455X (gfx1250) — matrix operand order repaired and checked by reading, not compiled, not run on hardware
//
#include <hip/hip_runtime.h>
#include <hip/hip_bf16.h>

// Problem sizes (match reference)
#define BATCH 1024
#define D0    2048
#define D1    2048
#define MM    16000
#define OUTN  3000
#define OUTP  3072   // OUT padded to multiple of 128 for block N-tiles

typedef __bf16 bf16_t;
typedef __attribute__((ext_vector_type(16))) __bf16 v16bf;
typedef __attribute__((ext_vector_type(8)))  __bf16 v8bf;
typedef __attribute__((ext_vector_type(8)))  float  v8f;
typedef __attribute__((ext_vector_type(4)))  unsigned int u32x4;
typedef __attribute__((ext_vector_type(8)))  int i32x8;
typedef __attribute__((ext_vector_type(4)))  int i32x4;

// ---------------- complex helpers ----------------
__device__ __forceinline__ float2 cmul(float2 a, float2 b) {
  return make_float2(fmaf(a.x, b.x, -a.y * b.y), fmaf(a.x, b.y, a.y * b.x));
}
__device__ __forceinline__ float2 cadd(float2 a, float2 b) {
  return make_float2(a.x + b.x, a.y + b.y);
}

// ---------------- Stockham mixed-radix FFT stages (forward, DIF, autosort) --
// Invariant: n * s == MM at every stage.  Butterflies per stage: (n/R)*s.
// y[q + s*(R*p + u)] = (sum_r x[q + s*(p + m*r)] * wR^{u*r}) * exp(-2*pi*i*p*u/n)

__device__ __forceinline__ void radix2_stage(const float2* src, float2* dst,
                                             int n, int s, int tid, int nthr) {
  const int m = n >> 1;
  const float ang = -6.28318530717958647692f / (float)n;
  const int total = m * s;
  for (int idx = tid; idx < total; idx += nthr) {
    const int q = idx % s;
    const int p = idx / s;
    const float2 a = src[q + s * p];
    const float2 b = src[q + s * (p + m)];
    const float2 sum = make_float2(a.x + b.x, a.y + b.y);
    const float2 dif = make_float2(a.x - b.x, a.y - b.y);
    float sn, cs;
    __sincosf(ang * (float)p, &sn, &cs);
    dst[q + s * (2 * p)]     = sum;
    dst[q + s * (2 * p + 1)] = cmul(dif, make_float2(cs, sn));
  }
}

__device__ __forceinline__ void radix5_stage(const float2* src, float2* dst,
                                             int n, int s, int tid, int nthr) {
  const int m = n / 5;
  const float ang = -6.28318530717958647692f / (float)n;
  // w5^k = exp(-2*pi*i*k/5)
  const float2 W5[5] = {
    make_float2( 1.0f,        0.0f),
    make_float2( 0.30901699f, -0.95105652f),
    make_float2(-0.80901699f, -0.58778525f),
    make_float2(-0.80901699f,  0.58778525f),
    make_float2( 0.30901699f,  0.95105652f)
  };
  const int total = m * s;
  for (int idx = tid; idx < total; idx += nthr) {
    const int q = idx % s;
    const int p = idx / s;
    float2 a[5];
    #pragma unroll
    for (int r = 0; r < 5; ++r) a[r] = src[q + s * (p + m * r)];
    #pragma unroll
    for (int u = 0; u < 5; ++u) {
      float2 acc = a[0];
      #pragma unroll
      for (int r = 1; r < 5; ++r) acc = cadd(acc, cmul(a[r], W5[(u * r) % 5]));
      float sn, cs;
      if (u == 0) { sn = 0.0f; cs = 1.0f; }
      else        __sincosf(ang * (float)(p * u), &sn, &cs);
      dst[q + s * (5 * p + u)] = cmul(acc, make_float2(cs, sn));
    }
  }
}

// Full 16000-point forward FFT: radices {5,5,5,2,2,2,2,2,2,2}, 10 stages.
// Even number of ping-pongs -> result ends up back in X.
__device__ __forceinline__ void fft16000(float2* X, float2* Y, int tid, int nthr) {
  float2* src = X;
  float2* dst = Y;
  int n = MM, s = 1;
  #pragma unroll 1
  for (int st = 0; st < 3; ++st) {
    radix5_stage(src, dst, n, s, tid, nthr);
    __syncthreads();
    n /= 5; s *= 5;
    float2* t = src; src = dst; dst = t;
  }
  #pragma unroll 1
  for (int st = 0; st < 7; ++st) {
    radix2_stage(src, dst, n, s, tid, nthr);
    __syncthreads();
    n >>= 1; s <<= 1;
    float2* t = src; src = dst; dst = t;
  }
  // result in X
}

// ---------------- Kernel 1: count-sketch + circular convolution per row -----
// One workgroup per batch row.  LDS: X[16000] complex + Y[16000] complex
// (ping-pong) = 250 KB dynamic LDS (<= 320 KB/WGP).
// p0 is packed into Re, p1 into Im; one FFT produces both spectra.
__global__ __launch_bounds__(256) void sketch_conv_kernel(
    const float* __restrict__ x0, const float* __restrict__ x1,
    const int* __restrict__ h0, const int* __restrict__ h1,
    const int* __restrict__ s0, const int* __restrict__ s1,
    __hip_bfloat16* __restrict__ zbf) {
  extern __shared__ float2 lds_fft[];
  float2* X = lds_fft;
  float2* Y = lds_fft + MM;
  const int tid  = threadIdx.x;
  const int nthr = blockDim.x;
  const int b    = blockIdx.x;

  for (int i = tid; i < MM; i += nthr) X[i] = make_float2(0.0f, 0.0f);
  __syncthreads();

  // count sketch via LDS float atomics: Re += sketch(x0), Im += sketch(x1)
  for (int j = tid; j < D0; j += nthr) {
    const float v0 = x0[(size_t)b * D0 + j] * (float)(2 * s0[j] - 1);
    atomicAdd(&X[h0[j]].x, v0);
    const float v1 = x1[(size_t)b * D1 + j] * (float)(2 * s1[j] - 1);
    atomicAdd(&X[h1[j]].y, v1);
  }
  __syncthreads();

  // forward FFT of (p0 + i*p1)
  fft16000(X, Y, tid, nthr);

  // Separate spectra, multiply, and store conj(Z) in place:
  //   C[k] = P0[k] + i*P1[k];  conj(C[N-k]) = P0[k] - i*P1[k]
  //   P0[k] = (C[k] + conj(C[N-k]))/2 ; P1[k] = -i*(C[k] - conj(C[N-k]))/2
  //   Z[k]  = P0[k]*P1[k];  array <- conj(Z)  (so ifft == forward fft, Re/N)
  for (int k = tid; k <= MM / 2; k += nthr) {
    const int kk = (k == 0) ? 0 : (MM - k);
    const float2 ck  = X[k];
    const float2 ckk = X[kk];
    const float2 p0 = make_float2(0.5f * (ck.x + ckk.x), 0.5f * (ck.y - ckk.y));
    const float2 p1 = make_float2(0.5f * (ck.y + ckk.y), 0.5f * (ckk.x - ck.x));
    const float2 z  = cmul(p0, p1);
    X[k]  = make_float2(z.x, -z.y);  // conj(Z)[k]
    X[kk] = z;                       // conj(Z)[N-k] = Z[k]
  }
  __syncthreads();

  // inverse FFT: ifft(Z) = conj(fft(conj(Z)))/N ; z is real -> take Re/N
  fft16000(X, Y, tid, nthr);

  const float invN = 1.0f / (float)MM;
  for (int i = tid; i < MM; i += nthr)
    zbf[(size_t)b * MM + i] = __float2bfloat16(X[i].x * invN);
}

// ---------------- Kernel 2: W f32 -> bf16, padded to 3072 rows --------------
__global__ __launch_bounds__(256) void convert_w_kernel(
    const float* __restrict__ W, __hip_bfloat16* __restrict__ Wbf) {
  const size_t idx = (size_t)blockIdx.x * blockDim.x + threadIdx.x;
  if (idx >= (size_t)OUTP * MM) return;
  const size_t row = idx / MM;
  const float v = (row < OUTN) ? W[idx] : 0.0f;
  Wbf[idx] = __float2bfloat16(v);
}

// ---------------- TDM: stage one 32(K) x 128(N) bf16 B-tile into LDS --------
// D# per CDNA5 ISA ch.8: 2-D tile, data_size=2B, tensor 16000 x 3072,
// dim0 stride 16000.  LDS padding: pad_interval code 3 (every 64 B) with
// pad_amount code 3 (16 B) -> 80 B row stride (bank-conflict-free fragment
// reads, 16 B aligned).
#define BROW_BYTES 80
#define BTILE_BYTES (128 * BROW_BYTES)   // 10240 B per buffer

__device__ __forceinline__ void tdm_load_b_tile(const bf16_t* gsrc,
                                                unsigned lds_off) {
  const unsigned long long ga = (unsigned long long)(uintptr_t)gsrc;
  u32x4 g0;
  g0[0] = 1u;                                   // count=1, user descriptor
  g0[1] = lds_off;                              // lds_addr (bytes)
  g0[2] = (unsigned)(ga & 0xffffffffu);         // global_addr[31:0]
  g0[3] = (unsigned)((ga >> 32) & 0x1ffffffu)   // global_addr[56:32]
        | 0x80000000u;                          // type=2 ("image")
  i32x8 g1;
  g1[0] = (int)(0x00010000u                     // data_size=1 (2 bytes)
        | (1u << 20)                            // pad_enable
        | (3u << 22)                            // pad_interval: 16 dwords (64B)
        | (3u << 25));                          // pad_amount: 4 dwords (16B)
  g1[1] = (int)(((unsigned)MM & 0xffffu) << 16);        // tensor_dim0[15:0]
  g1[2] = (int)((((unsigned)MM >> 16) & 0xffffu)        // tensor_dim0[31:16]
        | (((unsigned)OUTP & 0xffffu) << 16));          // tensor_dim1[15:0]
  g1[3] = (int)((((unsigned)OUTP >> 16) & 0xffffu)      // tensor_dim1[31:16]
        | (32u << 16));                                 // tile_dim0 = 32
  g1[4] = (int)128;                                     // tile_dim1 = 128
  g1[5] = (int)MM;                                      // dim0_stride[31:0]
  g1[6] = 0;
  g1[7] = 0;
  const i32x4 gz = {0, 0, 0, 0};
#if __clang_major__ >= 23
  const i32x8 gz8 = {0, 0, 0, 0, 0, 0, 0, 0};
  __builtin_amdgcn_tensor_load_to_lds(g0, g1, gz, gz, gz8, 0);
#else
  __builtin_amdgcn_tensor_load_to_lds(g0, g1, gz, gz, 0);
#endif
}

// ---------------- Kernel 3: bf16 WMMA GEMM + bias + ReLU --------------------
// out[m, n] = relu(sum_k z[m,k] * W[n,k] + bias[n])
// Block tile 128M x 128N: 8 waves arranged 4(M) x 2(N); each wave computes
// 32M x 64N = 2 A-fragments x 4 B-fragments -> 8 WMMAs per K-step, with each
// B fragment feeding two WMMAs (forces fragments into live registers, halves
// LDS read traffic vs a 16x128 wave tile).
// B strip (128 rows x 32 K) staged per K-step via TENSOR_LOAD_TO_LDS,
// double-buffered, shared by all 8 waves (8x L2-traffic reduction on W).
// A global loads software-pipelined one K-step ahead.
// Grid: (1024/128) x (3072/128) = 8 x 24 = 192 blocks.
__global__ __launch_bounds__(256) void gemm_wmma_kernel(
    const bf16_t* __restrict__ Z, const bf16_t* __restrict__ Wb,
    const float* __restrict__ bias, float* __restrict__ out) {
  __shared__ __align__(64) unsigned char bsm[2][BTILE_BYTES];

  const int lane  = threadIdx.x & 31;
  const int wave  = threadIdx.x >> 5;
  const int mBlk  = blockIdx.x / 24;
  const int nBlk  = blockIdx.x % 24;
  const int mBase = mBlk * 128 + (wave & 3) * 32;   // this wave's 32 M rows
  const int nBase = nBlk * 128;
  const int nSub  = (wave >> 2) * 64;               // this wave's 64 N cols
  const int half  = lane >> 4;                 // 0: lanes 0-15, 1: lanes 16-31
  const int l     = lane & 15;

  // A (16x32 bf16) lane layout: lanes 0-15 hold row M=l, K {0..7,16..23};
  // lanes 16-31 hold row M=l, K {8..15,24..31}.  Two fragments: rows +0, +16.
  const bf16_t* aPtr0 = Z + (size_t)(mBase + l) * MM + half * 8;
  const bf16_t* aPtr1 = aPtr0 + (size_t)16 * MM;
  // B fragment source in LDS: lane holds W row (nBase + nSub + t*16 + l),
  // K {half*8..half*8+7, 16+half*8..16+half*8+7} (80 B padded row stride).
  const unsigned bFragOff = (unsigned)((nSub + l) * BROW_BYTES + half * 16);

  v8f acc[2][4] = {};

  const int nSteps = MM / 32;                  // 500
  // prologue: stage first B tile; prefetch first A fragments
  if (wave == 0)
    tdm_load_b_tile(Wb + (size_t)nBase * MM, (unsigned)(uintptr_t)&bsm[0][0]);

  v16bf a_cur[2];
  #pragma unroll
  for (int ta = 0; ta < 2; ++ta) {
    const bf16_t* ap = (ta == 0) ? aPtr0 : aPtr1;
    const v8bf alo = *(const v8bf*)(ap);
    const v8bf ahi = *(const v8bf*)(ap + 16);
    #pragma unroll
    for (int j = 0; j < 8; ++j) { a_cur[ta][j] = alo[j]; a_cur[ta][j + 8] = ahi[j]; }
  }

  #pragma unroll 1
  for (int i = 0; i < nSteps; ++i) {
    const int buf = i & 1;
    if (wave == 0) __builtin_amdgcn_s_wait_tensorcnt(0);
    __syncthreads();   // buf ready for all; prior compute on buf^1 finished
    if (wave == 0 && (i + 1) < nSteps)
      tdm_load_b_tile(Wb + (size_t)nBase * MM + (size_t)(i + 1) * 32,
                      (unsigned)(uintptr_t)&bsm[buf ^ 1][0]);

    // prefetch next A fragments from global first (overlaps LDS + WMMA work);
    // last iteration re-reads current address (no branch, value unused)
    const int kn = ((i + 1) < nSteps) ? (i + 1) * 32 : i * 32;
    v16bf a_next[2];
    #pragma unroll
    for (int ta = 0; ta < 2; ++ta) {
      const bf16_t* ap = ((ta == 0) ? aPtr0 : aPtr1) + kn;
      const v8bf alo = *(const v8bf*)(ap);
      const v8bf ahi = *(const v8bf*)(ap + 16);
      #pragma unroll
      for (int j = 0; j < 8; ++j) { a_next[ta][j] = alo[j]; a_next[ta][j + 8] = ahi[j]; }
    }

    // load the wave's 4 B fragments; each feeds 2 WMMAs (stays in registers)
    const unsigned char* bufBase = &bsm[buf][0];
    v16bf bfrag[4];
    #pragma unroll
    for (int t = 0; t < 4; ++t) {
      const unsigned char* bp = bufBase + t * (16 * BROW_BYTES) + bFragOff;
      const v8bf blo = *(const v8bf*)(bp);
      const v8bf bhi = *(const v8bf*)(bp + 32);
      #pragma unroll
      for (int j = 0; j < 8; ++j) { bfrag[t][j] = blo[j]; bfrag[t][j + 8] = bhi[j]; }
    }

    #pragma unroll
    for (int ta = 0; ta < 2; ++ta) {
      #pragma unroll
      for (int t = 0; t < 4; ++t) {
        acc[ta][t] = __builtin_amdgcn_wmma_f32_16x16x32_bf16(
            /*neg_a=*/false, a_cur[ta], /*neg_b=*/false, bfrag[t],
            /*c_mod=*/(short)0, acc[ta][t], /*reuse_a=*/false, /*reuse_b=*/false);
      }
    }
    a_cur[0] = a_next[0];
    a_cur[1] = a_next[1];
  }

  // C/D layout: VGPR j -> M = (row base) + j, N = col base + l
  #pragma unroll
  for (int ta = 0; ta < 2; ++ta) {
    const int mRow = mBase + ta * 16 + half * 8;
    #pragma unroll
    for (int t = 0; t < 4; ++t) {
      const int n = nBase + nSub + t * 16 + l;
      if (n < OUTN) {
        const float bv = bias[n];
        #pragma unroll
        for (int j = 0; j < 8; ++j) {
          const float v = acc[ta][t][j] + bv;
          out[(size_t)(mRow + j) * OUTN + n] = v > 0.0f ? v : 0.0f;
        }
      }
    }
  }
}

// ---------------- launcher --------------------------------------------------
extern "C" void kernel_launch(void* const* d_in, const int* in_sizes, int n_in,
                              void* d_out, int out_size, void* d_ws, size_t ws_size,
                              hipStream_t stream) {
  const float* x0   = (const float*)d_in[0];
  const float* x1   = (const float*)d_in[1];
  const int*   h0   = (const int*)d_in[2];
  const int*   h1   = (const int*)d_in[3];
  const int*   s0   = (const int*)d_in[4];
  const int*   s1   = (const int*)d_in[5];
  const float* W    = (const float*)d_in[6];
  const float* bias = (const float*)d_in[7];
  float*       out  = (float*)d_out;

  // workspace: z_bf16 [1024 x 16000] then W_bf16 [3072 x 16000]
  __hip_bfloat16* zbf = (__hip_bfloat16*)d_ws;
  __hip_bfloat16* Wbf = zbf + (size_t)BATCH * MM;

  {
    const size_t total = (size_t)OUTP * MM;
    const int blocks = (int)((total + 255) / 256);
    convert_w_kernel<<<blocks, 256, 0, stream>>>(W, Wbf);
  }

  {
    const size_t ldsBytes = (size_t)2 * MM * sizeof(float2);  // 256000 B
    sketch_conv_kernel<<<BATCH, 256, ldsBytes, stream>>>(x0, x1, h0, h1, s0, s1, zbf);
  }

  gemm_wmma_kernel<<<8 * 24, 256, 0, stream>>>((const bf16_t*)zbf,
                                               (const bf16_t*)Wbf, bias, out);
}
